// CNN_62096637166429
// MI455X (gfx1250) — matrix operand order repaired and checked by reading, not compiled, not run on hardware
//
#include <hip/hip_runtime.h>
#include <cmath>

// ---------------- problem constants ----------------
#define Bb 8
#define Ll 2500
#define Vv 50000
#define Ee 100
#define Ff 256
#define Kk 9
#define Yy 8921

static constexpr int EP  = 128;           // E padded to 4x32 K-chunks
static constexpr int LP  = 2512;          // L padded to 16  (157 l-tiles)
static constexpr int LT  = 157;           // l tiles of 16
static constexpr int LP2 = 2528;          // L padded to 32  (79 K-chunks for m-GEMM)
static constexpr int KC  = 79;            // LP2/32
static constexpr int YT  = 558;           // ceil(Y/16)
static constexpr int YP  = 8928;          // YT*16

// d_out layout (floats): logits[B,Y], loss, alpha[B,Y,L], m[B,Y,F]
static constexpr size_t OUT_LOGITS = 0;
static constexpr size_t OUT_LOSS   = (size_t)Bb * Yy;                 // 71368
static constexpr size_t OUT_ALPHA  = OUT_LOSS + 1;                    // 71369
static constexpr size_t OUT_M      = OUT_ALPHA + (size_t)Bb * Yy * Ll;

// ---------------- WMMA fragment helpers (wave32, 16x16x32 bf16) ----------------
typedef __attribute__((ext_vector_type(16))) __bf16 v16bf;
typedef __attribute__((ext_vector_type(8)))  __bf16 v8bf;
typedef __attribute__((ext_vector_type(8)))  float  v8f;

__device__ __forceinline__ v8f wmma_bf16(v16bf a, v16bf b, v8f c) {
  return __builtin_amdgcn_wmma_f32_16x16x32_bf16(false, a, false, b, (short)0, c,
                                                 false, false);
}

// A-matrix 16x32 bf16, src row-major [M][K] with element stride `stride`.
// lane<16: M=lane, K={koff..koff+7, 16+koff..16+koff+7}, koff=0; lane>=16: koff=8.
__device__ __forceinline__ v16bf load_frag_a(const __bf16* src, int stride) {
  int lane = threadIdx.x & 31;
  int m    = lane & 15;
  int koff = (lane >> 4) << 3;
  const __bf16* p = src + (size_t)m * stride + koff;
  v8bf lo = *(const v8bf*)p;
  v8bf hi = *(const v8bf*)(p + 16);
  return __builtin_shufflevector(lo, hi, 0,1,2,3,4,5,6,7,8,9,10,11,12,13,14,15);
}

// B-matrix 32x16 bf16, src stored [N][K]-major with element stride `stride`.
// lane<16: N=lane, K={koff..koff+7, 16+koff..16+koff+7}, koff=0; lane>=16: N=lane-16, koff=8.
__device__ __forceinline__ v16bf load_frag_b(const __bf16* src, int stride) {
  int lane = threadIdx.x & 31;
  int n    = lane & 15;
  int koff = (lane >> 4) << 3;
  const __bf16* p = src + (size_t)n * stride + koff;
  v8bf lo = *(const v8bf*)p;
  v8bf hi = *(const v8bf*)(p + 16);
  return __builtin_shufflevector(lo, hi, 0,1,2,3,4,5,6,7,8,9,10,11,12,13,14,15);
}

// ---------------- preprocessing: f32 -> bf16 repacks ----------------
__global__ void k_cvt_emb(const float* __restrict__ W, __bf16* __restrict__ out) {
  int idx = blockIdx.x * 256 + threadIdx.x;
  if (idx >= Vv * EP) return;
  int v = idx >> 7, e = idx & 127;
  float x = (e < Ee) ? W[(size_t)v * Ee + e] : 0.f;
  out[idx] = (__bf16)x;
}

// conv_w [F][E][K] -> wT [k][f][e_pad] bf16
__global__ void k_cvt_w(const float* __restrict__ W, __bf16* __restrict__ out) {
  int idx = blockIdx.x * 256 + threadIdx.x;
  if (idx >= Kk * Ff * EP) return;
  int k = idx / (Ff * EP);
  int r = idx % (Ff * EP);
  int f = r >> 7, e = r & 127;
  float x = (e < Ee) ? W[((size_t)f * Ee + e) * Kk + k] : 0.f;
  out[idx] = (__bf16)x;
}

// U_w [Y][F] -> U_bf [YP][F] bf16 (zero padded rows)
__global__ void k_cvt_u(const float* __restrict__ W, __bf16* __restrict__ out) {
  int idx = blockIdx.x * 256 + threadIdx.x;
  if (idx >= YP * Ff) return;
  int y = idx >> 8, f = idx & 255;
  float x = (y < Yy) ? W[(size_t)y * Ff + f] : 0.f;
  out[idx] = (__bf16)x;
}

// ---------------- conv1d(K=9, same) + tanh via WMMA ----------------
// grid: B*LT blocks, 256 threads (8 waves). Each block: one (b, l-tile of 16).
// Each wave: two f-tiles of 16. GEMM: M=l(16), N=f(16), K = 9 taps x 128 e.
__global__ void k_conv(const int* __restrict__ tokens,
                       const float* __restrict__ conv_b,
                       const __bf16* __restrict__ embW,
                       const __bf16* __restrict__ wT,
                       __bf16* __restrict__ h_bf,
                       __bf16* __restrict__ hT_bf) {
  __shared__ __align__(16) __bf16 embTile[24 * EP];   // rows l0-4 .. l0+19
  int bx = blockIdx.x;
  int b  = bx / LT;
  int lt = bx % LT;
  int l0 = lt * 16;

  // cooperative gather: 24 rows x 128 e, 8-byte chunks
  const int* tok = tokens + (size_t)b * Ll;
  for (int c = threadIdx.x; c < 24 * 32; c += 256) {
    int r  = c >> 5;
    int e4 = (c & 31) << 2;
    int p  = l0 + r - 4;
    unsigned long long v = 0ull;
    if (p >= 0 && p < Ll) {
      int t = tok[p];
      v = *(const unsigned long long*)(embW + (size_t)t * EP + e4);
    }
    *(unsigned long long*)(embTile + r * EP + e4) = v;
  }
  __syncthreads();

  int lane = threadIdx.x & 31;
  int wave = threadIdx.x >> 5;
  int n    = lane & 15;
  int mo   = (lane >> 4) * 8;

  for (int tt = 0; tt < 2; ++tt) {
    int ft = wave * 2 + tt;
    v8f acc = {};
    for (int k = 0; k < Kk; ++k) {
      const __bf16* wb = wT + ((size_t)(k * Ff + ft * 16)) * EP;
      for (int ec = 0; ec < 4; ++ec) {
        v16bf a  = load_frag_a(embTile + k * EP + ec * 32, EP); // row m -> emb row m+k
        v16bf bf = load_frag_b(wb + ec * 32, EP);
        acc = wmma_bf16(a, bf, acc);
      }
    }
    int   f    = ft * 16 + n;
    float bias = conv_b[f];
    for (int i = 0; i < 8; ++i) {
      int    l  = l0 + mo + i;
      float  hv = tanhf(acc[i] + bias);
      __bf16 hb = (__bf16)hv;
      h_bf[((size_t)(b * LP + l)) * Ff + f]   = hb;  // [b][l][f]
      hT_bf[((size_t)(b * Ff + f)) * LP2 + l] = hb;  // [b][f][l]
    }
  }
}

// ---------------- fused attention: att -> softmax -> alpha -> m ----------------
// grid: B*YT blocks, 256 threads (8 waves). One (b, 16-row y-tile) per block.
// LDS: att f32 [16][2512] + alpha bf16 [16][2528] = 241,664 B (within 320KB WGP LDS)
__global__ void k_attn(const __bf16* __restrict__ U_bf,
                       const __bf16* __restrict__ h_bf,
                       const __bf16* __restrict__ hT_bf,
                       float* __restrict__ dout) {
  extern __shared__ char smem[];
  float*  attLds   = (float*)smem;
  __bf16* alphaLds = (__bf16*)(smem + (size_t)16 * LP * sizeof(float));

  int bx = blockIdx.x;
  int b  = bx / YT;
  int yt = bx % YT;
  int y0 = yt * 16;
  int lane = threadIdx.x & 31;
  int wave = threadIdx.x >> 5;
  int n    = lane & 15;
  int mo   = (lane >> 4) * 8;

  // U A-fragments held in registers: 8 K-chunks over F=256
  v16bf ufrag[8];
  for (int fc = 0; fc < 8; ++fc)
    ufrag[fc] = load_frag_a(U_bf + (size_t)y0 * Ff + fc * 32, Ff);

  // Phase A: att[y, l] = sum_f U[y,f] h[b,l,f]
  for (int lt = wave; lt < LT; lt += 8) {
    v8f acc = {};
    const __bf16* hb = h_bf + ((size_t)(b * LP + lt * 16)) * Ff;
    for (int fc = 0; fc < 8; ++fc) {
      v16bf bf = load_frag_b(hb + fc * 32, Ff);
      acc = wmma_bf16(ufrag[fc], bf, acc);
    }
    int l = lt * 16 + n;
    for (int i = 0; i < 8; ++i)
      attLds[(mo + i) * LP + l] = (l < Ll) ? acc[i] : -3.0e38f;
  }
  __syncthreads();

  // Phase B: row softmax (wave handles rows {wave, wave+8})
  for (int rr = 0; rr < 2; ++rr) {
    int    r   = wave + rr * 8;
    float* row = attLds + (size_t)r * LP;
    float  mx  = -3.0e38f;
    for (int l = lane; l < Ll; l += 32) mx = fmaxf(mx, row[l]);
    for (int o = 16; o; o >>= 1) mx = fmaxf(mx, __shfl_xor(mx, o, 32));
    float s = 0.f;
    for (int l = lane; l < Ll; l += 32) {
      float e = __expf(row[l] - mx);
      row[l]  = e;
      s += e;
    }
    for (int o = 16; o; o >>= 1) s += __shfl_xor(s, o, 32);
    float inv = 1.f / s;
    int   y   = y0 + r;
    float* aout = dout + OUT_ALPHA + ((size_t)b * Yy + y) * Ll;
    for (int l = lane; l < LP2; l += 32) {
      float a = (l < Ll) ? row[l] * inv : 0.f;
      alphaLds[(size_t)r * LP2 + l] = (__bf16)a;     // zero-padded K tail
      if (l < Ll && y < Yy) aout[l] = a;
    }
  }
  __syncthreads();

  // Phase C: m[y, f] = sum_l alpha[y,l] h[b,l,f]  (K = 2528, B from hT [f][l])
  for (int tt = 0; tt < 2; ++tt) {
    int ft = wave * 2 + tt;
    v8f acc = {};
    const __bf16* htb = hT_bf + ((size_t)(b * Ff + ft * 16)) * LP2;
    for (int kc = 0; kc < KC; ++kc) {
      v16bf a  = load_frag_a(alphaLds + kc * 32, LP2);
      v16bf bf = load_frag_b(htb + kc * 32, LP2);
      acc = wmma_bf16(a, bf, acc);
    }
    int f = ft * 16 + n;
    for (int i = 0; i < 8; ++i) {
      int y = y0 + mo + i;
      if (y < Yy) dout[OUT_M + ((size_t)b * Yy + y) * Ff + f] = acc[i];
    }
  }
}

// ---------------- logits + loss ----------------
__global__ void k_logits(const float* __restrict__ fw,
                         const float* __restrict__ fb,
                         float* __restrict__ dout) {
  int idx = blockIdx.x * 256 + threadIdx.x;
  if (idx >= Bb * Yy) return;
  int y = idx % Yy;
  const float* mr = dout + OUT_M + (size_t)idx * Ff;
  const float* wr = fw + (size_t)y * Ff;
  float s = fb[y];
  for (int f = 0; f < Ff; ++f) s += wr[f] * mr[f];
  dout[OUT_LOGITS + idx] = s;
}

__global__ void k_loss(const float* __restrict__ target, float* __restrict__ dout) {
  __shared__ float red[256];
  int   t = threadIdx.x;
  float s = 0.f;
  for (int i = t; i < Bb * Yy; i += 256) {
    float l  = dout[OUT_LOGITS + i];
    float tg = target[i];
    s += fmaxf(l, 0.f) + log1pf(__expf(-fabsf(l))) - tg * l;
  }
  red[t] = s;
  __syncthreads();
  for (int o = 128; o; o >>= 1) {
    if (t < o) red[t] += red[t + o];
    __syncthreads();
  }
  if (t == 0) dout[OUT_LOSS] = red[0] / (float)(Bb * Yy);
}

// ---------------- launch ----------------
extern "C" void kernel_launch(void* const* d_in, const int* in_sizes, int n_in,
                              void* d_out, int out_size, void* d_ws, size_t ws_size,
                              hipStream_t stream) {
  const int*   tokens  = (const int*)d_in[0];
  const float* target  = (const float*)d_in[1];
  const float* embed_W = (const float*)d_in[2];
  const float* conv_w  = (const float*)d_in[3];
  const float* conv_b  = (const float*)d_in[4];
  const float* U_w     = (const float*)d_in[5];
  const float* final_w = (const float*)d_in[6];
  const float* final_b = (const float*)d_in[7];
  float* dout = (float*)d_out;

  // workspace partition (bf16), all 16B aligned
  char* ws = (char*)d_ws;
  __bf16* embW_bf = (__bf16*)ws;                                   // V*128
  __bf16* wT_bf   = (__bf16*)(ws + (size_t)Vv * EP * 2);           // 9*256*128
  __bf16* U_bf    = (__bf16*)((char*)wT_bf + (size_t)Kk * Ff * EP * 2); // 8928*256
  __bf16* h_bf    = (__bf16*)((char*)U_bf + (size_t)YP * Ff * 2);  // B*2512*256
  __bf16* hT_bf   = (__bf16*)((char*)h_bf + (size_t)Bb * LP * Ff * 2); // B*256*2528

  k_cvt_emb<<<(Vv * EP + 255) / 256, 256, 0, stream>>>(embed_W, embW_bf);
  k_cvt_w<<<(Kk * Ff * EP + 255) / 256, 256, 0, stream>>>(conv_w, wT_bf);
  k_cvt_u<<<(YP * Ff + 255) / 256, 256, 0, stream>>>(U_w, U_bf);

  k_conv<<<Bb * LT, 256, 0, stream>>>(tokens, conv_b, embW_bf, wT_bf, h_bf, hT_bf);

  size_t smem_attn = (size_t)16 * LP * sizeof(float) + (size_t)16 * LP2 * 2;
  k_attn<<<Bb * YT, 256, smem_attn, stream>>>(U_bf, h_bf, hT_bf, dout);

  k_logits<<<(Bb * Yy + 255) / 256, 256, 0, stream>>>(final_w, final_b, dout);
  k_loss<<<1, 256, 0, stream>>>(target, dout);
}
